// UnifiedEquivariantGNN_51479478010568
// MI455X (gfx1250) — matrix operand order repaired and checked by reading, not compiled, not run on hardware
//
#include <hip/hip_runtime.h>
#include <hip/hip_bf16.h>
#include <math.h>

// ---------------- types ----------------
typedef __bf16 bf16;
typedef bf16  bf16x8  __attribute__((ext_vector_type(8)));
typedef bf16  bf16x16 __attribute__((ext_vector_type(16)));
typedef float floatx8 __attribute__((ext_vector_type(8)));

union V16 { bf16x16 v; bf16x8 h8[2]; };

#define WAIT_DS() asm volatile("s_wait_dscnt 0" ::: "memory")

// ---------------- problem constants ----------------
#define NNODE 50000
#define NEDGE 800000
#define DD 64

// workspace layout (bytes)
#define SAGG_BYTES (NNODE * DD * 4)          // 12,800,000
#define VAGG_BYTES (NNODE * 3 * 4)           // 600,000
#define ZERO_BYTES (SAGG_BYTES + VAGG_BYTES) // 13,400,000
#define W_OFF      13400064                  // 256B aligned

// bf16 weight offsets (in halves) inside W region; all (Nout x Kpad) row-major
#define OFF_NGW1T 0       // 64 x 96   (from ng_w1 65x64)
#define OFF_NGW2T 6144    // 64 x 64
#define OFF_EGW1T 10240   // 64 x 96
#define OFF_EGW2T 16384   // 64 x 64
#define OFF_MGW1T 20480   // 128 x 192
#define OFF_MGW2T 45056   // 64 x 128
#define OFF_UPWT  53248   // 64 x 64
#define W_TOTAL_HALVES 57344

__device__ __forceinline__ float silu_f(float x) { return x / (1.f + __expf(-x)); }

// ---------------- wave-level GEMM: D(16xN) = X(16xKPAD) * BT^T ----------------
// X: LDS, row-major 16 x KPAD bf16.  BT: global, row-major Nout x KPAD bf16.
// A fragment per ISA 16-bit A layout: lanes<16 hold K {k..k+7, k+16..k+23},
// lanes>=16 hold K {k+8..k+15, k+24..k+31}.
// B fragment: same K order as A (lanes<16 K {0..7,16..23}, lanes>=16 K {8..15,24..31}).
template<int KPAD, int NTILES>
__device__ __forceinline__ void wave_gemm(const bf16* X, const bf16* BT,
                                          floatx8* acc, int lane) {
  const int m  = lane & 15;
  const int hh = lane >> 4;
#pragma unroll
  for (int nt = 0; nt < NTILES; ++nt) {
    floatx8 c;
#pragma unroll
    for (int i = 0; i < 8; ++i) c[i] = 0.f;
    const bf16* brow = BT + (nt * 16 + m) * KPAD;
#pragma unroll
    for (int ks = 0; ks < KPAD / 32; ++ks) {
      V16 a, b;
      a.h8[0] = *(const bf16x8*)(X + m * KPAD + ks * 32 + hh * 8);
      a.h8[1] = *(const bf16x8*)(X + m * KPAD + ks * 32 + hh * 8 + 16);
      b.h8[0] = *(const bf16x8*)(brow + ks * 32 + hh * 8);
      b.h8[1] = *(const bf16x8*)(brow + ks * 32 + hh * 8 + 16);
      c = __builtin_amdgcn_wmma_f32_16x16x32_bf16(false, a.v, false, b.v,
                                                  (short)0, c, false, false);
    }
    acc[nt] = c;
  }
}

// bias (+optional SiLU) then store D-layout accumulators to LDS bf16 (16 x ldy)
template<int NTILES, bool SILU>
__device__ __forceinline__ void store_act(const floatx8* acc, const float* bias,
                                          bf16* Y, int ldy, int lane) {
  const int m  = lane & 15;
  const int hh = lane >> 4;
#pragma unroll
  for (int nt = 0; nt < NTILES; ++nt) {
    const int n = nt * 16 + m;
    const float bv = bias[n];
#pragma unroll
    for (int r = 0; r < 8; ++r) {
      float val = acc[nt][r] + bv;
      if (SILU) val = silu_f(val);
      Y[(r + 8 * hh) * ldy + n] = (bf16)val;
    }
  }
}

// ---------------- per-wave LDS slab for the edge kernel ----------------
struct __align__(16) WaveShm {
  bf16  XAH2[16 * 128];  // phase A: X input 16x96; phase B: H2 hidden 16x128
  bf16  H1[16 * 64];
  bf16  XC[16 * 192];    // concat [nf_i | nf_j | ef]
  int   rowIdx[16];
  int   colIdx[16];
  float evL[48];         // edge_vec_unit per edge (x,y,z)
  float sres[48];        // [0:16) res_ij, [16:32) res_ji, [32:48) cross_mag
  float coeff[16];
};

// one 65->64(SiLU)->64 MLP branch: gather + bf16 convert + 2 GEMMs -> XC column block
__device__ __forceinline__ void branchMLP(
    bf16* XA, bf16* H1, bf16* XCdst,
    const float* __restrict__ srcBase, const int* idxLds, bool useIdx, long e0,
    const float* scal, const bf16* W1T, const bf16* W2T,
    const float* __restrict__ b1, const float* __restrict__ b2, int lane)
{
  const int rowm = lane >> 1;         // 2 lanes per row
  const int c0   = (lane & 1) * 32;
  const long srcRow = useIdx ? (long)idxLds[rowm] : (e0 + rowm);
  const float4* p4 = (const float4*)(srcBase + srcRow * 64 + c0);
  bf16* dst = XA + rowm * 96 + c0;
#pragma unroll
  for (int j = 0; j < 8; ++j) {
    float4 a = p4[j];
    dst[j * 4 + 0] = (bf16)a.x;
    dst[j * 4 + 1] = (bf16)a.y;
    dst[j * 4 + 2] = (bf16)a.z;
    dst[j * 4 + 3] = (bf16)a.w;
  }
  // column 64 = geometric scalar, columns 65..95 = zero pad
  for (int idx = lane; idx < 512; idx += 32) {
    const int rr = idx >> 5;
    const int kk = idx & 31;
    XA[rr * 96 + 64 + kk] = (bf16)((kk == 0) ? scal[rr] : 0.f);
  }
  WAIT_DS();
  floatx8 acc[4];
  wave_gemm<96, 4>(XA, W1T, acc, lane);
  store_act<4, true>(acc, b1, H1, 64, lane);
  WAIT_DS();
  wave_gemm<64, 4>(H1, W2T, acc, lane);
  store_act<4, false>(acc, b2, XCdst, 192, lane);
}

// ---------------- weight prep: fp32 (K x N) -> bf16 transposed (N x Kpad) ----------------
__device__ void convT(const float* w, int K, int N, int Kpad, bf16* dst,
                      int tid, int stride) {
  const int total = N * Kpad;
  for (int i = tid; i < total; i += stride) {
    const int n = i / Kpad;
    const int k = i - n * Kpad;
    dst[i] = (bf16)((k < K) ? w[k * N + n] : 0.f);
  }
}

__global__ void prep_kernel(const float* ngw1, const float* ngw2,
                            const float* egw1, const float* egw2,
                            const float* mgw1, const float* mgw2,
                            const float* upw, bf16* W) {
  const int tid = blockIdx.x * blockDim.x + threadIdx.x;
  const int nth = gridDim.x * blockDim.x;
  convT(ngw1, 65, 64, 96,  W + OFF_NGW1T, tid, nth);
  convT(ngw2, 64, 64, 64,  W + OFF_NGW2T, tid, nth);
  convT(egw1, 65, 64, 96,  W + OFF_EGW1T, tid, nth);
  convT(egw2, 64, 64, 64,  W + OFF_EGW2T, tid, nth);
  convT(mgw1, 192, 128, 192, W + OFF_MGW1T, tid, nth);
  convT(mgw2, 128, 64, 128,  W + OFF_MGW2T, tid, nth);
  convT(upw,  64, 64, 64,    W + OFF_UPWT,  tid, nth);
}

// ---------------- edge kernel: 16 edges per wave, 4 waves per block ----------------
__global__ __launch_bounds__(128) void edge_kernel(
    const float* __restrict__ s, const float* __restrict__ v,
    const int* __restrict__ eidx, const float* __restrict__ eattr,
    const float* __restrict__ evu,
    const float* __restrict__ ng_b1, const float* __restrict__ ng_b2,
    const float* __restrict__ eg_b1, const float* __restrict__ eg_b2,
    const float* __restrict__ mg_b1, const float* __restrict__ mg_b2,
    const float* __restrict__ pe_w, const float* __restrict__ pe_b,
    const bf16* __restrict__ W,
    float* __restrict__ s_agg, float* __restrict__ v_agg)
{
  __shared__ WaveShm shm[4];
  const int wave = threadIdx.x >> 5;
  const int lane = threadIdx.x & 31;
  WaveShm& S = shm[wave];
  const long tile = (long)blockIdx.x * 4 + wave;
  if (tile >= NEDGE / 16) return;
  const long e0 = tile * 16;
  const int m  = lane & 15;
  const int hh = lane >> 4;

  const bf16* ngW1T = W + OFF_NGW1T;
  const bf16* ngW2T = W + OFF_NGW2T;
  const bf16* egW1T = W + OFF_EGW1T;
  const bf16* egW2T = W + OFF_EGW2T;
  const bf16* mgW1T = W + OFF_MGW1T;
  const bf16* mgW2T = W + OFF_MGW2T;

  // ---- phase 1: geometric scalars (lanes 0..15, one edge each) ----
  if (lane < 16) {
    const long e = e0 + lane;
    const int r = eidx[e];
    const int c = eidx[NEDGE + e];
    const float ex = evu[e * 3 + 0], ey = evu[e * 3 + 1], ez = evu[e * 3 + 2];
    const float vix = v[(long)r * 3 + 0], viy = v[(long)r * 3 + 1], viz = v[(long)r * 3 + 2];
    const float vjx = v[(long)c * 3 + 0], vjy = v[(long)c * 3 + 1], vjz = v[(long)c * 3 + 2];
    const float dij = vix * ex + viy * ey + viz * ez;
    const float dji = -(vjx * ex + vjy * ey + vjz * ez);
    const float cx = viy * vjz - viz * vjy;
    const float cy = viz * vjx - vix * vjz;
    const float cz = vix * vjy - viy * vjx;
    S.rowIdx[lane] = r;
    S.colIdx[lane] = c;
    S.evL[lane * 3 + 0] = ex; S.evL[lane * 3 + 1] = ey; S.evL[lane * 3 + 2] = ez;
    S.sres[lane]      = 1.f - dij;
    S.sres[16 + lane] = 1.f - dji;
    S.sres[32 + lane] = sqrtf(cx * cx + cy * cy + cz * cz);
  }
  WAIT_DS();

  // ---- phase 2: three 65->64->64 MLP branches into XC ----
  branchMLP(S.XAH2, S.H1, S.XC + 0,   s,     S.rowIdx, true,  e0, S.sres,
            ngW1T, ngW2T, ng_b1, ng_b2, lane);
  branchMLP(S.XAH2, S.H1, S.XC + 64,  s,     S.colIdx, true,  e0, S.sres + 16,
            ngW1T, ngW2T, ng_b1, ng_b2, lane);
  branchMLP(S.XAH2, S.H1, S.XC + 128, eattr, S.rowIdx, false, e0, S.sres + 32,
            egW1T, egW2T, eg_b1, eg_b2, lane);
  WAIT_DS();

  // ---- phase 3: message MLP 192 -> 128 (SiLU) -> 64 ----
  floatx8 acc8[8];
  wave_gemm<192, 8>(S.XC, mgW1T, acc8, lane);
  store_act<8, true>(acc8, mg_b1, S.XAH2, 128, lane);  // H2 reuses XA slab
  WAIT_DS();
  floatx8 acc4[4];
  wave_gemm<128, 4>(S.XAH2, mgW2T, acc4, lane);

  // ---- phase 4: scatter msg (atomic segment-sum) + coeff = msg . pe_w ----
  float cr[8] = {0.f, 0.f, 0.f, 0.f, 0.f, 0.f, 0.f, 0.f};
#pragma unroll
  for (int nt = 0; nt < 4; ++nt) {
    const int n = nt * 16 + m;
    const float bv = mg_b2[n];
    const float pw = pe_w[n];
#pragma unroll
    for (int r = 0; r < 8; ++r) {
      const float val = acc4[nt][r] + bv;
      const int M = r + 8 * hh;                 // edge within tile
      atomicAdd(&s_agg[(long)S.rowIdx[M] * 64 + n], val);
      cr[r] += val * pw;
    }
  }
  // reduce over the 16 lanes of each half (n dimension)
#pragma unroll
  for (int off = 8; off >= 1; off >>= 1)
#pragma unroll
    for (int r = 0; r < 8; ++r) cr[r] += __shfl_xor(cr[r], off, 32);
  if (m == 0) {
    const float pb = pe_b[0];
#pragma unroll
    for (int r = 0; r < 8; ++r) S.coeff[r + 8 * hh] = cr[r] + pb;
  }
  WAIT_DS();

  // ---- phase 5: v_msg scatter ----
  if (lane < 16) {
    const float cf = S.coeff[lane];
    const long r = S.rowIdx[lane];
    atomicAdd(&v_agg[r * 3 + 0], S.evL[lane * 3 + 0] * cf);
    atomicAdd(&v_agg[r * 3 + 1], S.evL[lane * 3 + 1] * cf);
    atomicAdd(&v_agg[r * 3 + 2], S.evL[lane * 3 + 2] * cf);
  }
}

// ---------------- node kernel: 16 nodes per wave ----------------
__global__ __launch_bounds__(128) void node_kernel(
    const float* __restrict__ s, const float* __restrict__ v,
    const float* __restrict__ s_agg, const float* __restrict__ v_agg,
    const bf16* __restrict__ upWT, const float* __restrict__ up_b,
    const float* __restrict__ ln_g, const float* __restrict__ ln_b,
    float* __restrict__ s_out, float* __restrict__ v_out)
{
  struct __align__(16) NodeShm {
    bf16  XS[16 * 64];
    float SN[16 * 64];
    float mu[16];
    float rs[16];
  };
  __shared__ NodeShm shm[4];
  const int wave = threadIdx.x >> 5;
  const int lane = threadIdx.x & 31;
  NodeShm& S = shm[wave];
  const long tile = (long)blockIdx.x * 4 + wave;
  if (tile >= NNODE / 16) return;
  const long n0 = tile * 16;
  const int m  = lane & 15;
  const int hh = lane >> 4;

  // silu(s_agg) -> bf16 LDS
  {
    const int rowm = lane >> 1;
    const int c0   = (lane & 1) * 32;
    const float4* p4 = (const float4*)(s_agg + (n0 + rowm) * 64 + c0);
    bf16* dst = S.XS + rowm * 64 + c0;
#pragma unroll
    for (int j = 0; j < 8; ++j) {
      float4 a = p4[j];
      dst[j * 4 + 0] = (bf16)silu_f(a.x);
      dst[j * 4 + 1] = (bf16)silu_f(a.y);
      dst[j * 4 + 2] = (bf16)silu_f(a.z);
      dst[j * 4 + 3] = (bf16)silu_f(a.w);
    }
  }
  WAIT_DS();
  floatx8 acc[4];
  wave_gemm<64, 4>(S.XS, upWT, acc, lane);

  // s_new = s + GEMM + up_b ; accumulate moments per row
  float s1[8] = {0,0,0,0,0,0,0,0};
  float s2[8] = {0,0,0,0,0,0,0,0};
#pragma unroll
  for (int nt = 0; nt < 4; ++nt) {
    const int n = nt * 16 + m;
    const float ub = up_b[n];
#pragma unroll
    for (int r = 0; r < 8; ++r) {
      const int M = r + 8 * hh;
      const long node = n0 + M;
      const float val = acc[nt][r] + ub + s[node * 64 + n];
      S.SN[M * 64 + n] = val;
      s1[r] += val;
      s2[r] += val * val;
    }
  }
#pragma unroll
  for (int off = 8; off >= 1; off >>= 1)
#pragma unroll
    for (int r = 0; r < 8; ++r) {
      s1[r] += __shfl_xor(s1[r], off, 32);
      s2[r] += __shfl_xor(s2[r], off, 32);
    }
  if (m == 0) {
#pragma unroll
    for (int r = 0; r < 8; ++r) {
      const float mean = s1[r] * (1.f / 64.f);
      const float var  = s2[r] * (1.f / 64.f) - mean * mean;
      S.mu[r + 8 * hh] = mean;
      S.rs[r + 8 * hh] = rsqrtf(var + 1e-5f);
    }
  }
  WAIT_DS();
  for (int idx = lane; idx < 1024; idx += 32) {
    const int M = idx >> 6;
    const int n = idx & 63;
    const long node = n0 + M;
    s_out[node * 64 + n] = (S.SN[idx] - S.mu[M]) * S.rs[M] * ln_g[n] + ln_b[n];
  }
  if (lane < 16) {
    const long node = n0 + lane;
    const float a = v[node * 3 + 0] + v_agg[node * 3 + 0];
    const float b = v[node * 3 + 1] + v_agg[node * 3 + 1];
    const float c = v[node * 3 + 2] + v_agg[node * 3 + 2];
    float nr = sqrtf(a * a + b * b + c * c);
    nr = fmaxf(nr, 1e-6f);
    v_out[node * 3 + 0] = a / nr;
    v_out[node * 3 + 1] = b / nr;
    v_out[node * 3 + 2] = c / nr;
  }
}

// ---------------- host launch ----------------
extern "C" void kernel_launch(void* const* d_in, const int* in_sizes, int n_in,
                              void* d_out, int out_size, void* d_ws, size_t ws_size,
                              hipStream_t stream) {
  const float* s     = (const float*)d_in[0];
  const float* v     = (const float*)d_in[1];
  const int*   eidx  = (const int*)  d_in[2];
  const float* eattr = (const float*)d_in[3];
  const float* evu   = (const float*)d_in[4];
  const float* ng_w1 = (const float*)d_in[5];
  const float* ng_b1 = (const float*)d_in[6];
  const float* ng_w2 = (const float*)d_in[7];
  const float* ng_b2 = (const float*)d_in[8];
  const float* eg_w1 = (const float*)d_in[9];
  const float* eg_b1 = (const float*)d_in[10];
  const float* eg_w2 = (const float*)d_in[11];
  const float* eg_b2 = (const float*)d_in[12];
  const float* mg_w1 = (const float*)d_in[13];
  const float* mg_b1 = (const float*)d_in[14];
  const float* mg_w2 = (const float*)d_in[15];
  const float* mg_b2 = (const float*)d_in[16];
  const float* pe_w  = (const float*)d_in[17];
  const float* pe_b  = (const float*)d_in[18];
  const float* up_w  = (const float*)d_in[19];
  const float* up_b  = (const float*)d_in[20];
  const float* ln_g  = (const float*)d_in[21];
  const float* ln_b  = (const float*)d_in[22];

  float* s_agg = (float*)d_ws;
  float* v_agg = (float*)((char*)d_ws + SAGG_BYTES);
  bf16*  W     = (bf16*)((char*)d_ws + W_OFF);

  hipMemsetAsync(d_ws, 0, ZERO_BYTES, stream);
  prep_kernel<<<64, 256, 0, stream>>>(ng_w1, ng_w2, eg_w1, eg_w2, mg_w1, mg_w2, up_w, W);

  edge_kernel<<<(NEDGE / 16 + 3) / 4, 128, 0, stream>>>(
      s, v, eidx, eattr, evu,
      ng_b1, ng_b2, eg_b1, eg_b2, mg_b1, mg_b2, pe_w, pe_b,
      W, s_agg, v_agg);

  float* s_out = (float*)d_out;
  float* v_out = s_out + (long)NNODE * DD;
  node_kernel<<<(NNODE / 16 + 3) / 4, 128, 0, stream>>>(
      s, v, s_agg, v_agg, W + OFF_UPWT, up_b, ln_g, ln_b, s_out, v_out);
}
